// GCNEncoder_22823456211322
// MI455X (gfx1250) — matrix operand order repaired and checked by reading, not compiled, not run on hardware
//
#include <hip/hip_runtime.h>
#include <hip/hip_bf16.h>

typedef __attribute__((ext_vector_type(16))) __bf16 v16bf;
typedef __attribute__((ext_vector_type(8)))  float  v8f;

#define IN_CH  256
#define HID_CH 128
#define OUT_CH 64

// ---------------------------------------------------------------------------
// Pack fp32 A[M][K] (row-major) into WMMA A-fragment order, bf16, optional ReLU.
// Packed index: ((rt*KT + kt)*32 + lane)*16 + i   (KT = K/32 k-tiles)
//   lane<16 : M=rt*16+lane,    i<8 -> K=kt*32+i,          i>=8 -> K=kt*32+16+(i-8)
//   lane>=16: M=rt*16+lane-16, i<8 -> K=kt*32+8+i,        i>=8 -> K=kt*32+24+(i-8)
// Writes are fully coalesced; each lane's 16 values are 32B-contiguous for the GEMM.
// ---------------------------------------------------------------------------
template <int K>
__global__ void pack_A_bf16(const float* __restrict__ src, __bf16* __restrict__ dstp,
                            int M, int Mpad, int relu) {
    constexpr int KT = K / 32;
    long o = (long)blockIdx.x * blockDim.x + threadIdx.x;
    if (o >= (long)Mpad * K) return;
    int  i    = (int)(o & 15);
    long t    = o >> 4;
    int  lane = (int)(t & 31);
    long tt   = t >> 5;
    int  kt   = (int)(tt % KT);
    int  rt   = (int)(tt / KT);
    int  half = lane >> 4;
    int  row  = rt * 16 + (lane & 15);
    if (row >= M) row = M - 1;                       // clamp (pad rows replicate)
    int  k = kt * 32 + (i < 8 ? half * 8 + i : 16 + half * 8 + (i - 8));
    float v = src[(long)row * K + k];
    if (relu) v = fmaxf(v, 0.0f);
    dstp[o] = (__bf16)v;
}

// ---------------------------------------------------------------------------
// Pack fp32 B[K][N] (row-major) into WMMA B-fragment order, bf16.
// Packed index: ((nt*KT + kt)*32 + lane)*16 + i
//   value = B[kt*32 + (i<8 ? (lane>>4)*8 + i : 16 + (lane>>4)*8 + (i-8))][nt*16 + (lane&15)]
// ---------------------------------------------------------------------------
template <int K, int N>
__global__ void pack_B_bf16(const float* __restrict__ src, __bf16* __restrict__ dstp) {
    constexpr int KT = K / 32;
    int o = blockIdx.x * blockDim.x + threadIdx.x;
    if (o >= K * N) return;
    int i    = o & 15;
    int t    = o >> 4;
    int lane = t & 31;
    int tt   = t >> 5;
    int kt   = tt % KT;
    int nt   = tt / KT;
    int k = kt * 32 + (i < 8 ? (lane >> 4) * 8 + i : 16 + (lane >> 4) * 8 + (i - 8));
    int n = nt * 16 + (lane & 15);
    dstp[o] = (__bf16)src[k * N + n];
}

// ---------------------------------------------------------------------------
// WMMA GEMM on pre-packed operands:  C[M][N] = A[M][K] * B[K][N], C fp32.
// Block = (N/16) waves; wave w owns col-tile w, block owns row-tile blockIdx.x.
// Inner loop per k-tile: one 32B A load + one 32B B load + one v_wmma.
// ---------------------------------------------------------------------------
template <int K, int N>
__global__ void gemm_packed_wmma(const __bf16* __restrict__ Ap,
                                 const __bf16* __restrict__ Bp,
                                 float* __restrict__ C, int M) {
    constexpr int KT = K / 32;
    const int lane = threadIdx.x & 31;
    const int wave = threadIdx.x >> 5;       // col tile
    const int rt   = blockIdx.x;             // row tile

    const __bf16* a_ptr = Ap + ((long)rt   * KT * 32 + lane) * 16;
    const __bf16* b_ptr = Bp + ((long)wave * KT * 32 + lane) * 16;

    v8f acc = {};
#pragma unroll
    for (int kt = 0; kt < KT; ++kt) {
        v16bf a = *(const v16bf*)(a_ptr + (long)kt * 32 * 16);
        v16bf b = *(const v16bf*)(b_ptr + (long)kt * 32 * 16);
        acc = __builtin_amdgcn_wmma_f32_16x16x32_bf16(
            /*neg_a=*/false, a, /*neg_b=*/false, b,
            /*c_mod=*/(short)0, acc, /*reuse_a=*/false, /*reuse_b=*/false);
    }

    const int half = lane >> 4, l15 = lane & 15;
    const int m0 = rt * 16, n0 = wave * 16;
    if (m0 + 16 <= M) {                       // full tile: unconditional stores
#pragma unroll
        for (int r = 0; r < 8; ++r)
            C[(long)(m0 + half * 8 + r) * N + n0 + l15] = acc[r];
    } else {
#pragma unroll
        for (int r = 0; r < 8; ++r) {
            int row = m0 + half * 8 + r;
            if (row < M) C[(long)row * N + n0 + l15] = acc[r];
        }
    }
}

// ---------------------------------------------------------------------------
// degree / normalization
// ---------------------------------------------------------------------------
__global__ void init_deg(float* __restrict__ deg, int n) {
    int i = blockIdx.x * blockDim.x + threadIdx.x;
    if (i < n) deg[i] = 1.0f;                // self-loop contribution
}

__global__ void count_deg(const long long* __restrict__ dst, float* __restrict__ deg, int e) {
    int i = blockIdx.x * blockDim.x + threadIdx.x;
    if (i < e) atomicAdd(&deg[(int)dst[i]], 1.0f);
}

__global__ void rsqrt_deg(const float* __restrict__ deg, float* __restrict__ dinv, int n) {
    int i = blockIdx.x * blockDim.x + threadIdx.x;
    if (i < n) dinv[i] = rsqrtf(deg[i]);     // deg >= 1 always (self-loops)
}

// ---------------------------------------------------------------------------
// agg[i][c] = dinv[i]^2 * H[i][c] + bias[c]  (self-loop term + bias folded in)
// ---------------------------------------------------------------------------
template <int C>
__global__ void init_agg(const float* __restrict__ H, const float* __restrict__ dinv,
                         const float* __restrict__ bias, float* __restrict__ agg, int N) {
    long i = (long)blockIdx.x * blockDim.x + threadIdx.x;
    if (i < (long)N * C) {
        int node = (int)(i / C);
        int c    = (int)(i - (long)node * C);
        float dv = dinv[node];
        agg[i] = dv * dv * H[i] + bias[c];
    }
}

// ---------------------------------------------------------------------------
// one wave per edge: agg[dst] += dinv[src]*dinv[dst] * H[src]
// vectorized gather (float4 / float2); scalar global_atomic_add_f32 scatter
// ---------------------------------------------------------------------------
template <int C>
__global__ void edge_agg(const long long* __restrict__ src, const long long* __restrict__ dst,
                         const float* __restrict__ dinv, const float* __restrict__ H,
                         float* __restrict__ agg, int E) {
    constexpr int V = C / 32;                 // floats per lane (4 or 2)
    int w    = (int)(((long)blockIdx.x * blockDim.x + threadIdx.x) >> 5);
    int lane = threadIdx.x & 31;
    if (w >= E) return;
    int s = (int)src[w];
    int d = (int)dst[w];
    float nrm = dinv[s] * dinv[d];
    const float* hs = H   + (long)s * C + lane * V;
    float*       ad = agg + (long)d * C + lane * V;
    if constexpr (V == 4) {
        float4 h = *(const float4*)hs;
        atomicAdd(&ad[0], nrm * h.x);
        atomicAdd(&ad[1], nrm * h.y);
        atomicAdd(&ad[2], nrm * h.z);
        atomicAdd(&ad[3], nrm * h.w);
    } else {
        float2 h = *(const float2*)hs;
        atomicAdd(&ad[0], nrm * h.x);
        atomicAdd(&ad[1], nrm * h.y);
    }
}

// ---------------------------------------------------------------------------
extern "C" void kernel_launch(void* const* d_in, const int* in_sizes, int n_in,
                              void* d_out, int out_size, void* d_ws, size_t ws_size,
                              hipStream_t stream) {
    const float*     x   = (const float*)d_in[0];
    const long long* ei  = (const long long*)d_in[1];
    const float*     W1  = (const float*)d_in[2];
    const float*     b1  = (const float*)d_in[3];
    const float*     W2  = (const float*)d_in[4];
    const float*     b2  = (const float*)d_in[5];
    float*           out = (float*)d_out;

    const int  N    = in_sizes[0] / IN_CH;     // 50000
    const int  E    = in_sizes[1] / 2;         // 800000
    const int  Mpad = ((N + 15) / 16) * 16;    // row tiles (50000 is exact)
    const long long* srcI = ei;
    const long long* dstI = ei + E;

    // ---- workspace layout (explicit reuse, ~77 MB total) ----
    char*   ws   = (char*)d_ws;
    size_t  nP   = (size_t)Mpad;
    __bf16* Ap1  = (__bf16*)ws;                                    // Mpad*256*2
    float*  H1   = (float*)(ws + nP * IN_CH * 2);                  // N*128*4
    float*  agg1 = (float*)(ws + nP * IN_CH * 2 + (size_t)N * HID_CH * 4);
    char*   tail = ws + nP * IN_CH * 2 + 2 * (size_t)N * HID_CH * 4;
    __bf16* Bp1  = (__bf16*)tail;                                  // 256*128*2
    __bf16* Bp2  = (__bf16*)(tail + IN_CH * HID_CH * 2);           // 128*64*2
    float*  deg  = (float*)(tail + IN_CH * HID_CH * 2 + HID_CH * OUT_CH * 2);
    float*  dinv = deg + (size_t)N;
    // Ap1 region is dead after GEMM1 -> reuse:
    __bf16* Ap2  = (__bf16*)ws;                                    // Mpad*128*2
    float*  H2   = (float*)(ws + nP * HID_CH * 2);                 // N*64*4

    const int T = 256;
    auto blk = [](long n, int t) { return (int)((n + t - 1) / t); };

    // 1) pack operands into WMMA fragment layouts (bf16; x pass is the big one)
    pack_A_bf16<IN_CH><<<blk((long)Mpad * IN_CH, T), T, 0, stream>>>(x, Ap1, N, Mpad, 0);
    pack_B_bf16<IN_CH, HID_CH><<<blk(IN_CH * HID_CH, T), T, 0, stream>>>(W1, Bp1);
    pack_B_bf16<HID_CH, OUT_CH><<<blk(HID_CH * OUT_CH, T), T, 0, stream>>>(W2, Bp2);

    // 2) symmetric normalization terms
    init_deg<<<blk(N, T), T, 0, stream>>>(deg, N);
    count_deg<<<blk(E, T), T, 0, stream>>>(dstI, deg, E);
    rsqrt_deg<<<blk(N, T), T, 0, stream>>>(deg, dinv, N);

    // 3) layer 1: H1 = x @ W1   (packed bf16 WMMA, f32 accumulate)
    gemm_packed_wmma<IN_CH, HID_CH>
        <<<Mpad / 16, (HID_CH / 16) * 32, 0, stream>>>(Ap1, Bp1, H1, N);

    // 4) layer-1 aggregation: agg1 = b1 + dinv^2*H1 + sum_edges
    init_agg<HID_CH><<<blk((long)N * HID_CH, T), T, 0, stream>>>(H1, dinv, b1, agg1, N);
    edge_agg<HID_CH><<<blk((long)E * 32, T), T, 0, stream>>>(srcI, dstI, dinv, H1, agg1, E);

    // 5) ReLU + bf16 + fragment-pack in one pass (feeds layer-2 WMMA)
    pack_A_bf16<HID_CH><<<blk((long)Mpad * HID_CH, T), T, 0, stream>>>(agg1, Ap2, N, Mpad, 1);

    // 6) layer 2: H2 = relu(agg1) @ W2
    gemm_packed_wmma<HID_CH, OUT_CH>
        <<<Mpad / 16, (OUT_CH / 16) * 32, 0, stream>>>(Ap2, Bp2, H2, N);

    // 7) layer-2 aggregation straight into d_out (init fully writes d_out first)
    init_agg<OUT_CH><<<blk((long)N * OUT_CH, T), T, 0, stream>>>(H2, dinv, b2, out, N);
    edge_agg<OUT_CH><<<blk((long)E * 32, T), T, 0, stream>>>(srcI, dstI, dinv, H2, out, E);
}
